// MultiHeadAttention_35364760715918
// MI455X (gfx1250) — matrix operand order repaired and checked by reading, not compiled, not run on hardware
//
#include <hip/hip_runtime.h>

// ---------------------------------------------------------------------------
// CDNA5 (gfx1250) multi-head attention forward.
//   B=2, S=4096, D=1024, H=16, hd=64, causal, fp32 in/out.
//   bf16 WMMA (v_wmma_f32_16x16x32_bf16), fp32 accumulation.
//   bf16 intermediate workspaces; async global->LDS staging for bf16 tiles.
// ---------------------------------------------------------------------------

typedef __attribute__((ext_vector_type(16))) __bf16 bf16x16;
typedef __attribute__((ext_vector_type(8)))  float  f32x8;

#define WMMA_BF16(a, b, c)                                                    \
  __builtin_amdgcn_wmma_f32_16x16x32_bf16(false, (a), false, (b), (short)0,   \
                                          (c), false, false)

// Pack two floats into two bf16 lanes of a dword (lowers to v_cvt_pk_bf16_f32)
__device__ __forceinline__ unsigned pk2(float a, float b) {
  union { __bf16 h[2]; unsigned u; } q;
  q.h[0] = (__bf16)a; q.h[1] = (__bf16)b;
  return q.u;
}

// Async DMA: 16 bytes global -> LDS, tracked by ASYNCcnt (gfx1250).
__device__ __forceinline__ void async_ld_b128(const void* g, void* lds) {
  const unsigned l = (unsigned)(uintptr_t)lds;  // low 32 bits = LDS byte addr
  asm volatile("global_load_async_to_lds_b128 %0, %1, off"
               :: "v"(l), "v"(g) : "memory");
}
__device__ __forceinline__ void wait_async() {
#if __has_builtin(__builtin_amdgcn_s_wait_asynccnt)
  __builtin_amdgcn_s_wait_asynccnt(0);
#else
  asm volatile("s_wait_asynccnt 0" ::: "memory");
#endif
}

// --- A-fragment (16x32 bf16, lane = row) from LDS row-major tile -----------
__device__ __forceinline__ bf16x16 ldsA(const __bf16* p, int ld, int lane) {
  const int m = lane & 15, half = lane >> 4;
  const __bf16* q0 = p + m * ld + half * 16;
  bf16x16 r;
#pragma unroll
  for (int i = 0; i < 8; ++i) {
    r[i]     = q0[i];
    r[i + 8] = q0[8 + i];
  }
  return r;
}

// --- B-fragment (32x16 bf16, lane = col) from LDS tile stored [n][k] -------
__device__ __forceinline__ bf16x16 ldsB(const __bf16* p, int ld, int lane) {
  const int n = lane & 15, half = lane >> 4;
  const __bf16* q0 = p + n * ld + half * 16;
  bf16x16 r;
#pragma unroll
  for (int i = 0; i < 16; ++i) r[i] = q0[i];
  return r;
}

// ---------------------------------------------------------------------------
// Tiled GEMM: C[M,N] = A[M,K] @ Bw[K,N] + bias[N]
//   A_BF16:  A is bf16 (async copy staging) vs fp32 (convert staging)
//   OUT_BF16: C stored as bf16 (optionally scaled for n < scale_cols)
// Block tile 128x128, K-step 64. 8 waves, wave w owns rows 16w..16w+15.
// ---------------------------------------------------------------------------
template <bool A_BF16, bool OUT_BF16>
__global__ __launch_bounds__(256) void gemm_bias_kern(
    const void* __restrict__ Ap, const float* __restrict__ Bw,
    const float* __restrict__ bias, void* __restrict__ Cp,
    int M, int N, int K, int scale_cols, float scale) {
  __shared__ __bf16 As[128][72];   // 128 x 64 (+8 pad)
  __shared__ __bf16 Bt[128][72];   // [n][k]: transposed 64 x 128 tile

  const int tid  = threadIdx.x;
  const int lane = tid & 31;
  const int w    = tid >> 5;
  const int m0   = blockIdx.y * 128;
  const int n0   = blockIdx.x * 128;

  f32x8 acc[8] = {};

  for (int k0 = 0; k0 < K; k0 += 64) {
    if (k0 + 64 < K)
      __builtin_prefetch(Bw + (size_t)(k0 + 64 + (tid >> 3)) * N + n0, 0, 1);

    // ---- Stage A tile (128 x 64) ----
    if constexpr (A_BF16) {
      const __bf16* Ah = (const __bf16*)Ap;
#pragma unroll
      for (int i = 0; i < 4; ++i) {            // 128 rows x 8 chunks of 16B
        const int idx = tid + i * 256;
        const int r = idx >> 3, c = (idx & 7) * 8;
        async_ld_b128(Ah + (size_t)(m0 + r) * K + k0 + c, &As[r][c]);
      }
    } else {
      const float* Af = (const float*)Ap;
#pragma unroll
      for (int i = 0; i < 8; ++i) {            // 2048 float4's
        const int idx = tid + i * 256;
        const int r = idx >> 4, c = (idx & 15) << 2;
        const float4 v = *(const float4*)(Af + (size_t)(m0 + r) * K + k0 + c);
        uint2 u; u.x = pk2(v.x, v.y); u.y = pk2(v.z, v.w);
        *(uint2*)&As[r][c] = u;                // ds_store_b64
      }
    }

    // ---- Stage W tile transposed via column gather: Bt[n][k] ----
#pragma unroll
    for (int i = 0; i < 8; ++i) {              // 128 n x 16 k-quads
      const int idx = tid + i * 256;
      const int n = idx & 127, kq = idx >> 7;
      const float* col = Bw + (size_t)(k0 + kq * 4) * N + n0 + n;
      const float f0 = col[0];
      const float f1 = col[(size_t)N];
      const float f2 = col[(size_t)2 * N];
      const float f3 = col[(size_t)3 * N];
      uint2 u; u.x = pk2(f0, f1); u.y = pk2(f2, f3);
      *(uint2*)&Bt[n][kq * 4] = u;             // ds_store_b64
    }

    if constexpr (A_BF16) wait_async();
    __syncthreads();

#pragma unroll
    for (int ks = 0; ks < 64; ks += 32) {
      const bf16x16 a = ldsA(&As[w * 16][0] + ks, 72, lane);
#pragma unroll
      for (int t = 0; t < 8; ++t) {
        const bf16x16 b = ldsB(&Bt[t * 16][0] + ks, 72, lane);
        acc[t] = WMMA_BF16(a, b, acc[t]);
      }
    }
    __syncthreads();
  }

  // ---- Epilogue: bias (+ optional scale) + store ----
  const int half = lane >> 4, nl = lane & 15;
#pragma unroll
  for (int t = 0; t < 8; ++t) {
    const int n = n0 + t * 16 + nl;
    const float bv = bias[n];
    const float sc = (n < scale_cols) ? scale : 1.0f;
#pragma unroll
    for (int g = 0; g < 8; ++g) {
      const int m = m0 + w * 16 + g + 8 * half;
      const float v = (acc[t][g] + bv) * sc;
      if constexpr (OUT_BF16)
        ((__bf16*)Cp)[(size_t)m * N + n] = (__bf16)v;
      else
        ((float*)Cp)[(size_t)m * N + n] = v;
    }
  }
}

// ---------------------------------------------------------------------------
// Flash attention forward (causal). qkv: bf16 [B, S, 3*D]; head h of Q/K/V at
// col which*1024 + h*64. Q is pre-scaled by 1/sqrt(hd) in the QKV epilogue.
// Block: 128 queries of one (b,h); 8 waves; key blocks of 64. Output: bf16.
// ---------------------------------------------------------------------------
__global__ __launch_bounds__(256) void attn_fwd(
    const __bf16* __restrict__ qkv, __bf16* __restrict__ outp, int Sl) {
  __shared__ __bf16 Qs[128][72];      // queries x head_dim
  __shared__ __bf16 Ks[64][72];       // keys x head_dim ([n][k] for B-frag)
  __shared__ __bf16 Vt[64][72];       // head_dim x keys ([n][k] for B-frag)
  __shared__ __bf16 Ps[8][16][72];    // per-wave P re-layout patch

  const int tid = threadIdx.x, lane = tid & 31, w = tid >> 5;
  const int qb = blockIdx.x;
  const int b = blockIdx.y >> 4, h = blockIdx.y & 15;
  const size_t rs = 3072;             // qkv row stride (bf16 elems)
  const __bf16* base = qkv + (size_t)b * Sl * rs + h * 64;
  const int q0 = qb * 128;
  const int half = lane >> 4, nl = lane & 15;
  const int qrow_base = q0 + w * 16 + 8 * half;   // + g -> global query row

  // Stage Q via async DMA (128 rows x 128B); completion waited in first iter.
#pragma unroll
  for (int i = 0; i < 4; ++i) {
    const int idx = tid + i * 256;
    const int r = idx >> 3, c = (idx & 7) * 8;
    async_ld_b128(base + (size_t)(q0 + r) * rs + c, &Qs[r][c]);
  }

  f32x8 oacc[4] = {};
  float mst[8], lst[8];
#pragma unroll
  for (int g = 0; g < 8; ++g) { mst[g] = -1e30f; lst[g] = 0.0f; }

  const int nkb = 2 * qb + 2;     // causal: keys 0 .. q0+127

  for (int kb = 0; kb < nkb; ++kb) {
    // Stage K rows via async DMA (64 rows x 128B)
#pragma unroll
    for (int i = 0; i < 2; ++i) {
      const int idx = tid + i * 256;
      const int r = idx >> 3, c = (idx & 7) * 8;
      async_ld_b128(base + (size_t)(kb * 64 + r) * rs + 1024 + c, &Ks[r][c]);
    }
    // Stage V transposed: Vt[d][key] (bf16 source, 4 elems per thread-iter)
#pragma unroll
    for (int i = 0; i < 4; ++i) {
      const int idx = tid + i * 256;            // 64 rows x 16 d-quads
      const int r = idx >> 4, c = (idx & 15) << 2;
      union { uint2 u; __bf16 h[4]; } vv;
      vv.u = *(const uint2*)(base + (size_t)(kb * 64 + r) * rs + 2048 + c);
      Vt[c + 0][r] = vv.h[0]; Vt[c + 1][r] = vv.h[1];
      Vt[c + 2][r] = vv.h[2]; Vt[c + 3][r] = vv.h[3];
    }
    wait_async();
    __syncthreads();

    // S = Q K^T  (16 query rows x 64 key cols per wave)
    f32x8 sacc[4] = {};
#pragma unroll
    for (int ks = 0; ks < 64; ks += 32) {
      const bf16x16 aq = ldsA(&Qs[w * 16][0] + ks, 72, lane);
#pragma unroll
      for (int t = 0; t < 4; ++t) {
        const bf16x16 bk = ldsB(&Ks[t * 16][0] + ks, 72, lane);
        sacc[t] = WMMA_BF16(aq, bk, sacc[t]);
      }
    }

    // Causal mask + online softmax (C-frag: row = g + 8*half, col = nl)
#pragma unroll
    for (int g = 0; g < 8; ++g) {
      const int qrow = qrow_base + g;
      float sv[4];
      float mx = -1e30f;
#pragma unroll
      for (int t = 0; t < 4; ++t) {
        const int kcol = kb * 64 + t * 16 + nl;
        float s = sacc[t][g];
        s = (kcol <= qrow) ? s : -1e30f;
        sv[t] = s;
        mx = fmaxf(mx, s);
      }
#pragma unroll
      for (int d = 1; d < 16; d <<= 1) mx = fmaxf(mx, __shfl_xor(mx, d, 32));
      const float mnew = fmaxf(mst[g], mx);
      const float corr = __expf(mst[g] - mnew);
      float rsum = 0.0f;
#pragma unroll
      for (int t = 0; t < 4; ++t) {
        const float p = __expf(sv[t] - mnew);
        rsum += p;
        Ps[w][g + 8 * half][t * 16 + nl] = (__bf16)p;   // C-layout -> LDS
      }
#pragma unroll
      for (int d = 1; d < 16; d <<= 1) rsum += __shfl_xor(rsum, d, 32);
      lst[g] = lst[g] * corr + rsum;
      mst[g] = mnew;
#pragma unroll
      for (int t = 0; t < 4; ++t) oacc[t][g] *= corr;
    }

    // O += P V  (same-wave DS ops are in-order; per-wave Ps patch needs no
    // extra barrier)
#pragma unroll
    for (int ks = 0; ks < 64; ks += 32) {
      const bf16x16 ap = ldsA(&Ps[w][0][0] + ks, 72, lane);
#pragma unroll
      for (int t = 0; t < 4; ++t) {
        const bf16x16 bv = ldsB(&Vt[t * 16][0] + ks, 72, lane);
        oacc[t] = WMMA_BF16(ap, bv, oacc[t]);
      }
    }
    __syncthreads();
  }

  // Epilogue: normalize, store bf16 to [B, S, H*hd]
  __bf16* orow = outp + (size_t)b * Sl * 1024 + h * 64;
#pragma unroll
  for (int g = 0; g < 8; ++g) {
    const float inv = 1.0f / lst[g];
    const int qrow = qrow_base + g;
#pragma unroll
    for (int t = 0; t < 4; ++t) {
      orow[(size_t)qrow * 1024 + t * 16 + nl] = (__bf16)(oacc[t][g] * inv);
    }
  }
}

// ---------------------------------------------------------------------------
extern "C" void kernel_launch(void* const* d_in, const int* in_sizes, int n_in,
                              void* d_out, int out_size, void* d_ws,
                              size_t ws_size, hipStream_t stream) {
  const float* x     = (const float*)d_in[0];
  const float* w_qkv = (const float*)d_in[1];
  const float* b_qkv = (const float*)d_in[2];
  const float* w_out = (const float*)d_in[3];
  const float* b_out = (const float*)d_in[4];
  float* out = (float*)d_out;

  const int B = 2, S = 4096, D = 1024;
  const int M = B * S;                          // 8192

  __bf16* qkv  = (__bf16*)d_ws;                 // [M, 3*D] bf16 (50.3 MB)
  __bf16* attn = qkv + (size_t)M * 3 * D;       // [M, D]   bf16 (16.8 MB)

  // 1) QKV projection -> bf16, with 1/sqrt(hd) folded into Q columns.
  dim3 g1(3 * D / 128, M / 128);
  gemm_bias_kern<false, true><<<g1, 256, 0, stream>>>(
      x, w_qkv, b_qkv, qkv, M, 3 * D, D, /*scale_cols=*/D, /*scale=*/0.125f);

  // 2) Causal flash attention per (b, h) -> bf16.
  dim3 g2(S / 128, B * 16);
  attn_fwd<<<g2, 256, 0, stream>>>(qkv, attn, S);

  // 3) Output projection -> fp32 result.
  dim3 g3(D / 128, M / 128);
  gemm_bias_kern<true, false><<<g3, 256, 0, stream>>>(
      attn, w_out, b_out, out, M, D, D, /*scale_cols=*/0, /*scale=*/1.0f);
}
